// Critic_15607911153666
// MI455X (gfx1250) — matrix operand order repaired and checked by reading, not compiled, not run on hardware
//
#include <hip/hip_runtime.h>
#include <hip/hip_bf16.h>
#include <math.h>

// Problem constants (from reference): B=1024, T=512, D=64, H=128, 4H=512
#define B_TOT   1024
#define T_STEPS 512
#define D_IN    64
#define H_DIM   128
#define G4      512
#define BTILE   16
#define THREADS 256

// LDS padded strides (elements); strides stay off 64-bank multiples.
#define GATE_STRIDE 520   // [16][G4] f32  (2080B rows)
#define H_STRIDE    136   // [16][H]  bf16 (272B rows)
#define X_STRIDE    72    // [16][D]  bf16 (144B rows)

// LDS byte offsets (all 16B aligned). Weights now live in VGPRs, not LDS.
#define LDS_GATES 0
#define LDS_H     (LDS_GATES + BTILE*GATE_STRIDE*4)   // 33280
#define LDS_X     (LDS_H + BTILE*H_STRIDE*2)          // 37632
#define LDS_QP    (LDS_X + BTILE*X_STRIDE*2)          // 39936
#define LDS_TOTAL (LDS_QP + THREADS*4)                // 40960 bytes

typedef __attribute__((ext_vector_type(16))) __bf16 v16bf;
typedef __attribute__((ext_vector_type(8)))  float  v8f;

union FragU  { v16bf v; uint4 q[2]; };
union Pack8  { uint4 q; unsigned short us[8]; };
union Pack4  { uint2 q; unsigned short us[4]; };

// gfx1250 hardware transcendentals (guarded; fall back to libm if absent)
#if __has_builtin(__builtin_amdgcn_tanhf)
#define DEV_TANH(x) __builtin_amdgcn_tanhf(x)
#elif __has_builtin(__builtin_amdgcn_tanh_f32)
#define DEV_TANH(x) __builtin_amdgcn_tanh_f32(x)
#else
#define DEV_TANH(x) tanhf(x)
#endif
#if __has_builtin(__builtin_amdgcn_exp2f)
#define DEV_EXP2(x) __builtin_amdgcn_exp2f(x)
#else
#define DEV_EXP2(x) exp2f(x)
#endif
#if __has_builtin(__builtin_amdgcn_rcpf)
#define DEV_RCP(x) __builtin_amdgcn_rcpf(x)
#else
#define DEV_RCP(x) (1.0f/(x))
#endif

__device__ __forceinline__ unsigned short f32_to_bf16(float x) {
    unsigned int u = __float_as_uint(x);
    u += 0x7FFFu + ((u >> 16) & 1u);        // round-to-nearest-even
    return (unsigned short)(u >> 16);
}

__device__ __forceinline__ float sigmoidf_(float x) {
    // 1/(1+2^(-x*log2e)) : v_exp_f32 is base-2 on AMD; v_rcp beats v_div.
    return DEV_RCP(1.0f + DEV_EXP2(x * -1.44269504088896f));
}

// ---------------------------------------------------------------------------
// Prep: W_ih, W_hh -> bf16 (row-major, same layout), bias = b_ih + b_hh (f32)
// ---------------------------------------------------------------------------
__global__ void critic_prep_kernel(const float* __restrict__ wih,
                                   const float* __restrict__ whh,
                                   const float* __restrict__ bih,
                                   const float* __restrict__ bhh,
                                   unsigned short* __restrict__ wih_bf,
                                   unsigned short* __restrict__ whh_bf,
                                   float* __restrict__ bias) {
    const int idx = blockIdx.x * blockDim.x + threadIdx.x;
    if (idx < G4 * D_IN)  wih_bf[idx] = f32_to_bf16(wih[idx]);
    if (idx < G4 * H_DIM) whh_bf[idx] = f32_to_bf16(whh[idx]);
    if (idx < G4)         bias[idx]   = bih[idx] + bhh[idx];
}

// ---------------------------------------------------------------------------
// Fused LSTM scan + head. One workgroup owns 16 batch rows for all 512 steps.
// All 24 weight B-fragments (192 VGPRs) are register-resident across the scan
// (CDNA5 wave32 supports up to 1024 VGPRs; launch_bounds gives 2 waves/SIMD).
// ---------------------------------------------------------------------------
__global__ void __launch_bounds__(THREADS, 1)
critic_lstm_kernel(const float* __restrict__ state,   // [B,T,D]
                   const float* __restrict__ action,  // [B,1]
                   const unsigned short* __restrict__ wih_bf, // [G4,D] bf16
                   const unsigned short* __restrict__ whh_bf, // [G4,H] bf16
                   const float* __restrict__ bias,    // [G4] = b_ih+b_hh
                   const float* __restrict__ fcs1_W,  // [H]
                   const float* __restrict__ fcs1_b,  // [1]
                   const float* __restrict__ fca1_W,  // [1]
                   const float* __restrict__ fca1_b,  // [1]
                   float* __restrict__ out) {         // [B]
    extern __shared__ char lds[];
    float*          sGates = (float*)(lds + LDS_GATES);
    unsigned short* sH     = (unsigned short*)(lds + LDS_H);
    unsigned short* sX     = (unsigned short*)(lds + LDS_X);
    float*          sQ     = (float*)(lds + LDS_QP);

    const int tid  = threadIdx.x;
    const int wave = tid >> 5;
    const int lane = tid & 31;
    const int r0   = blockIdx.x * BTILE;

    // h0 = 0
    for (int i = tid; i < BTILE * H_STRIDE; i += THREADS) sH[i] = 0;

    // Elementwise role: thread -> (row, 8 h-columns); c lives in registers.
    const int ew_row  = tid >> 4;
    const int ew_col0 = (tid & 15) * 8;
    float c_reg[8], h_reg[8];
#pragma unroll
    for (int j = 0; j < 8; ++j) { c_reg[j] = 0.f; h_reg[j] = 0.f; }

    // WMMA role constants (ISA 7.12.2 layouts, wave32).
    const int lm   = lane & 15;       // M row (A) / N col (B/C)
    const int hi   = lane >> 4;       // lane half
    const int a_k0 = hi * 8;          // A frag: 8-elt chunks at K=a_k0, a_k0+16
    const int b_k0 = hi * 8;          // B frag: 8-elt chunks at K=b_k0, b_k0+16

    // ---- One-time: pull all weight B-fragments + bias into registers ----
    // Frag k index: 0..1 = W_ih k-steps (K=64), 2..5 = W_hh k-steps (K=128).
    v16bf wB[24];
    float bsum[4];
#pragma unroll
    for (int nt = 0; nt < 4; ++nt) {
        const int ncol = (wave * 4 + nt) * 16 + lm;
        bsum[nt] = bias[ncol];
#pragma unroll
        for (int kk = 0; kk < 2; ++kk) {
            const unsigned short* p = wih_bf + ncol * D_IN + kk * 32 + b_k0;
            FragU u; u.q[0] = *(const uint4*)p; u.q[1] = *(const uint4*)(p + 16);
            wB[nt * 6 + kk] = u.v;
        }
#pragma unroll
        for (int kk = 0; kk < 4; ++kk) {
            const unsigned short* p = whh_bf + ncol * H_DIM + kk * 32 + b_k0;
            FragU u; u.q[0] = *(const uint4*)p; u.q[1] = *(const uint4*)(p + 16);
            wB[nt * 6 + 2 + kk] = u.v;
        }
    }

    const float* xbase = state + (size_t)r0 * T_STEPS * D_IN;
    const int xr = tid >> 4, xc = (tid & 15) * 4;  // x staging role

    __syncthreads();

#pragma unroll 1
    for (int t = 0; t < T_STEPS; ++t) {
        // ---- Stage x_t tile [16][64] f32 -> bf16 LDS (one b64 store) ----
        {
            const float* src = xbase + (size_t)xr * T_STEPS * D_IN
                                     + (size_t)t * D_IN + xc;
            const float4 v = *(const float4*)src;
            Pack4 p4;
            p4.us[0] = f32_to_bf16(v.x); p4.us[1] = f32_to_bf16(v.y);
            p4.us[2] = f32_to_bf16(v.z); p4.us[3] = f32_to_bf16(v.w);
            *(uint2*)(sX + xr * X_STRIDE + xc) = p4.q;
            if (t + 1 < T_STEPS)
                __builtin_prefetch(src + D_IN, 0, 0);  // global_prefetch_b8
        }
        __syncthreads();  // [A] sX(t) and sH(t-1) visible

        // ---- A fragments, shared by this wave's 4 N-tiles ----
        v16bf xA[2], hA[4];
#pragma unroll
        for (int kk = 0; kk < 2; ++kk) {
            const unsigned short* p = sX + lm * X_STRIDE + kk * 32 + a_k0;
            FragU u; u.q[0] = *(const uint4*)p; u.q[1] = *(const uint4*)(p + 16);
            xA[kk] = u.v;
        }
#pragma unroll
        for (int kk = 0; kk < 4; ++kk) {
            const unsigned short* p = sH + lm * H_STRIDE + kk * 32 + a_k0;
            FragU u; u.q[0] = *(const uint4*)p; u.q[1] = *(const uint4*)(p + 16);
            hA[kk] = u.v;
        }

        // ---- gates = bias + x@Wih^T + h@Whh^T (all weights in VGPRs) ----
#pragma unroll
        for (int nt = 0; nt < 4; ++nt) {
            v8f acc;
#pragma unroll
            for (int r = 0; r < 8; ++r) acc[r] = bsum[nt];
#pragma unroll
            for (int kk = 0; kk < 2; ++kk)
                acc = __builtin_amdgcn_wmma_f32_16x16x32_bf16(
                          false, xA[kk], false, wB[nt * 6 + kk],
                          (short)0, acc, false, false);
#pragma unroll
            for (int kk = 0; kk < 4; ++kk)
                acc = __builtin_amdgcn_wmma_f32_16x16x32_bf16(
                          false, hA[kk], false, wB[nt * 6 + 2 + kk],
                          (short)0, acc, false, false);
            // C/D layout: VGPR r, half 'hi' -> row m = r + 8*hi, col = lm
            float* gdst = sGates + (hi * 8) * GATE_STRIDE + (wave * 4 + nt) * 16 + lm;
#pragma unroll
            for (int r = 0; r < 8; ++r) gdst[r * GATE_STRIDE] = acc[r];
        }
        __syncthreads();  // [B] gates(t) visible

        // ---- Elementwise LSTM cell (i,f,g,o), vectorized LDS access ----
        {
            const float* grow = sGates + ew_row * GATE_STRIDE + ew_col0;
            float ig[8], fg[8], gg[8], og[8];
            *(float4*)&ig[0] = *(const float4*)(grow);
            *(float4*)&ig[4] = *(const float4*)(grow + 4);
            *(float4*)&fg[0] = *(const float4*)(grow + 128);
            *(float4*)&fg[4] = *(const float4*)(grow + 132);
            *(float4*)&gg[0] = *(const float4*)(grow + 256);
            *(float4*)&gg[4] = *(const float4*)(grow + 260);
            *(float4*)&og[0] = *(const float4*)(grow + 384);
            *(float4*)&og[4] = *(const float4*)(grow + 388);
            Pack8 hp;
#pragma unroll
            for (int j = 0; j < 8; ++j) {
                float c = sigmoidf_(fg[j]) * c_reg[j]
                        + sigmoidf_(ig[j]) * DEV_TANH(gg[j]);
                c_reg[j] = c;
                const float h = sigmoidf_(og[j]) * DEV_TANH(c);
                h_reg[j] = h;
                hp.us[j] = f32_to_bf16(h);
            }
            *(uint4*)(sH + ew_row * H_STRIDE + ew_col0) = hp.q;
        }
        // no third barrier: next iteration's barrier [A] orders sH w->r
    }

    // ---- Head: q = hT @ fcs1_W^T + fcs1_b + action*fca1_W + fca1_b ----
    float part = 0.f;
#pragma unroll
    for (int j = 0; j < 8; ++j) part += h_reg[j] * fcs1_W[ew_col0 + j];
    sQ[tid] = part;
    __syncthreads();
    if (tid < BTILE) {
        float s = 0.f;
#pragma unroll
        for (int cb = 0; cb < 16; ++cb) s += sQ[tid * 16 + cb];
        const int row = r0 + tid;
        out[row] = s + fcs1_b[0] + action[row] * fca1_W[0] + fca1_b[0];
    }
}

// ---------------------------------------------------------------------------
extern "C" void kernel_launch(void* const* d_in, const int* in_sizes, int n_in,
                              void* d_out, int out_size, void* d_ws, size_t ws_size,
                              hipStream_t stream) {
    (void)in_sizes; (void)n_in; (void)out_size; (void)ws_size;
    const float* state  = (const float*)d_in[0];
    const float* action = (const float*)d_in[1];
    const float* W_ih   = (const float*)d_in[2];
    const float* W_hh   = (const float*)d_in[3];
    const float* b_ih   = (const float*)d_in[4];
    const float* b_hh   = (const float*)d_in[5];
    const float* fcs1_W = (const float*)d_in[6];
    const float* fcs1_b = (const float*)d_in[7];
    const float* fca1_W = (const float*)d_in[8];
    const float* fca1_b = (const float*)d_in[9];
    float* out = (float*)d_out;

    // Workspace: [0,64K) Wih bf16, [64K,192K) Whh bf16, [192K,+2K) bias f32
    char* ws = (char*)d_ws;
    unsigned short* wih_bf = (unsigned short*)(ws);
    unsigned short* whh_bf = (unsigned short*)(ws + (size_t)G4 * D_IN * 2);
    float* bias = (float*)(ws + (size_t)G4 * D_IN * 2 + (size_t)G4 * H_DIM * 2);

    critic_prep_kernel<<<(G4 * H_DIM + THREADS - 1) / THREADS, THREADS, 0, stream>>>(
        W_ih, W_hh, b_ih, b_hh, wih_bf, whh_bf, bias);

    (void)hipFuncSetAttribute((const void*)critic_lstm_kernel,
                              hipFuncAttributeMaxDynamicSharedMemorySize,
                              LDS_TOTAL);

    critic_lstm_kernel<<<B_TOT / BTILE, THREADS, LDS_TOTAL, stream>>>(
        state, action, wih_bf, whh_bf, bias,
        fcs1_W, fcs1_b, fca1_W, fca1_b, out);
}
